// BiasedAxialAttention_13932873908961
// MI455X (gfx1250) — compile-verified, matrix operand order checked by reading, not run on hardware
//
#include <hip/hip_runtime.h>
#include <hip/hip_bf16.h>
#include <math.h>

// ---------------------------------------------------------------------------
// BiasedAxialAttention for gfx1250 (MI455X).  bf16 WMMA (16x16x32) everywhere,
// f32 accumulate.  Q/K staging buffers (113 MB) kept resident in 192 MB L2.
// All WMMA operands have the contraction axis contiguous -> every fragment is
// two b128 loads per lane.  Shared operands (Q strip in the score GEMM, the
// V tile in the output GEMM) are staged into LDS with the Tensor Data Mover
// (TENSOR_LOAD_TO_LDS, double-buffered, s_wait_tensorcnt) so waves don't
// re-load identical fragments from L2.
// ---------------------------------------------------------------------------

typedef __attribute__((ext_vector_type(16))) __bf16 v16bf;
typedef __attribute__((ext_vector_type(8)))  float  v8f;
typedef __attribute__((ext_vector_type(4)))  unsigned uint4v;
typedef __attribute__((ext_vector_type(4)))  unsigned uint32x4;
typedef __attribute__((ext_vector_type(8)))  int      int32x8;
typedef __attribute__((ext_vector_type(4)))  int      int32x4;

#define LSEQ 384
#define DE   128
#define DB   64
#define NH   4
#define DH   48
#define DM   192                    // NH*DH
#define NK   (LSEQ*DH)              // 18432  score contraction dim
#define NI   (LSEQ*LSEQ)            // 147456 flattened (n,i) rows
#define HQ   ((size_t)LSEQ*NK)      // per-head Q/K elements
#define HV   ((size_t)LSEQ*LSEQ*DH) // per-head V elements
#define SCALE_Q 0.14433756729740643f // 1/sqrt(48)
#define INV_L   (1.0f/384.0f)
#define INF_M   1000000.0f

// ---- workspace layout (bytes), every offset 16B aligned --------------------
static constexpr size_t SZ_EL   = (size_t)NI*DE*2;      // LN(edge^T) bf16 [r][k]
static constexpr size_t SZ_WCAT = (size_t)768*128*2;    // [q|k|v|g]^T  [c][k]
static constexpr size_t SZ_WOB  = (size_t)128*192*2;    // Wo^T         [c][k]
static constexpr size_t SZ_QK   = (size_t)NH*HQ*2;      // [h][i][n*48+d]
static constexpr size_t SZ_V    = (size_t)NH*HV*2;      // [h][n][d][j]
static constexpr size_t SZ_G    = (size_t)NI*DM*2;      // gate  [n*L+i][hc]
static constexpr size_t SZ_BH   = (size_t)NH*NI*4;      // b_h   [h][i][j] f32
static constexpr size_t SZ_P    = (size_t)NH*NI*2;      // attn  [h][i][j]
static constexpr size_t SZ_O    = (size_t)NI*DM*2;      // gated out [r][hc]

static constexpr size_t OFF_EL   = 0;
static constexpr size_t OFF_WCAT = OFF_EL   + SZ_EL;
static constexpr size_t OFF_WOB  = OFF_WCAT + SZ_WCAT;
static constexpr size_t OFF_Q    = OFF_WOB  + SZ_WOB;
static constexpr size_t OFF_K    = OFF_Q    + SZ_QK;
static constexpr size_t OFF_V    = OFF_K    + SZ_QK;
static constexpr size_t OFF_G    = OFF_V    + SZ_V;
static constexpr size_t OFF_BH   = OFF_G    + SZ_G;
static constexpr size_t OFF_P    = OFF_BH   + SZ_BH;
static constexpr size_t OFF_O    = OFF_P    + SZ_P;   // total ~325 MB

// ---- bf16 <-> f32 helpers (bit-level storage as ushort) --------------------
__device__ __forceinline__ unsigned short f2b(float f) {
    union { __bf16 h; unsigned short s; } u; u.h = (__bf16)f; return u.s;
}
__device__ __forceinline__ float b2f(unsigned short s) {
    union { __bf16 h; unsigned short s2; } u; u.s2 = s; return (float)u.h;
}

union Frag { v16bf v; uint4v q[2]; unsigned u[8]; unsigned short s[16]; };

// Fragment loader, contraction axis contiguous in memory (global or LDS):
// element (M|N = lane&15, K) at base[(row0 + lane&15)*ld + k0 + K].
// Per ISA 7.12.2: lanes 0-15 hold K-base 0, lanes 16-31 K-base 8;
// VGPR v<4 -> K = kb+2v, v>=4 -> K = kb+16+2(v-4).
// Each lane's data = two contiguous 16-byte runs -> two *_load_b128.
__device__ __forceinline__ void load_frag(Frag& f, const unsigned short* base,
                                          int ld, int row0, int k0, int lane) {
    int r  = row0 + (lane & 15);
    int kb = (lane < 16) ? 0 : 8;
    const unsigned short* p = base + (size_t)r * ld + k0 + kb;
    f.q[0] = *reinterpret_cast<const uint4v*>(p);        // K = kb   .. kb+7
    f.q[1] = *reinterpret_cast<const uint4v*>(p + 16);   // K = kb+16.. kb+23
}

#define WMMA_BF16(accv, av, bv) \
    (accv) = __builtin_amdgcn_wmma_f32_16x16x32_bf16(false, (av), false, (bv), \
                                                     (short)0, (accv), false, false)

__device__ __forceinline__ v8f zero_v8f() {
    v8f z;
#pragma unroll
    for (int e = 0; e < 8; ++e) z[e] = 0.0f;
    return z;
}

// ---- Tensor Data Mover: 2-D tile (bf16) global -> LDS ----------------------
// D# per ISA ch.8: group0 = {count=1, lds_addr, global_addr[56:0], type=2},
// group1 = {data_size=2B, tensor_dim0/1 = tile dims (no OOB), tile_dim0/1,
// tensor_dim0_stride = real row stride (may exceed dim, per spec)}.
// Groups 2/3 zero (<=2-D tensor).  This toolchain: 6-arg builtin
// (uint32x4, int32x8, int32x4, int32x4, int32x8, i32 cpol).
__device__ __forceinline__ unsigned lds_addr_of(const void* p) {
    // low 32 bits of a flat shared-aperture address = LDS byte offset
    return (unsigned)(uintptr_t)p;
}
__device__ __forceinline__ void tdm_load_2d(unsigned lds_off, const void* gptr,
                                            unsigned tile_k, unsigned tile_rows,
                                            unsigned long long row_stride_elems) {
    uint32x4 g0 = (uint32x4)0u;
    int32x8  g1 = (int32x8)0;
    int32x4  g2 = (int32x4)0;
    int32x4  g3 = (int32x4)0;
    int32x8  g4 = (int32x8)0;
    unsigned long long ga = (unsigned long long)(uintptr_t)gptr;
    g0[0] = 1u;                                            // count=1 (valid D#)
    g0[1] = lds_off;                                       // lds_addr
    g0[2] = (unsigned)(ga & 0xFFFFFFFFu);                  // global_addr[31:0]
    g0[3] = (unsigned)((ga >> 32) & 0x01FFFFFFu) | (2u << 30);  // [56:32]|type=2
    g1[0] = (int)(1u << 16);                               // data_size: 2 bytes
    g1[1] = (int)((tile_k & 0xFFFFu) << 16);               // tensor_dim0 lo
    g1[2] = (int)(((tile_k >> 16) & 0xFFFFu) |
                  ((tile_rows & 0xFFFFu) << 16));          // td0 hi | td1 lo
    g1[3] = (int)(((tile_rows >> 16) & 0xFFFFu) |
                  ((tile_k & 0xFFFFu) << 16));             // td1 hi | tile_dim0
    g1[4] = (int)(tile_rows & 0xFFFFu);                    // tile_dim1 (dim2=0)
    g1[5] = (int)(row_stride_elems & 0xFFFFFFFFull);       // dim0_stride lo
    g1[6] = (int)((row_stride_elems >> 32) & 0xFFFFull);   // dim0_stride hi
    __builtin_amdgcn_tensor_load_to_lds(g0, g1, g2, g3, g4, 0);
}

// ---------------------------------------------------------------------------
// K0: pack weights to bf16, TRANSPOSED so K is contiguous.
// ---------------------------------------------------------------------------
__global__ void k_prep_weights(const float* Wq, const float* Wk, const float* Wv,
                               const float* Wg, const float* Wo,
                               unsigned short* wcatT, unsigned short* wobT) {
    int t = blockIdx.x * blockDim.x + threadIdx.x;
    if (t < 768*128) {
        int c = t / 128, k = t % 128;
        const float* W = (c < 192) ? Wq : (c < 384) ? Wk : (c < 576) ? Wv : Wg;
        wcatT[t] = f2b(W[k*192 + (c % 192)]);
    } else {
        int t2 = t - 768*128;
        if (t2 < 128*192) {
            int c = t2 / 192, k = t2 % 192;
            wobT[t2] = f2b(Wo[k*128 + c]);
        }
    }
}

// ---------------------------------------------------------------------------
// K1: eL[n][i][c] = LayerNorm(edge[i][n][0:128]) in bf16.  One wave per row.
// ---------------------------------------------------------------------------
__global__ void k_ln_edge(const float* __restrict__ edge, const float* g,
                          const float* bta, unsigned short* eL) {
    int w = threadIdx.x >> 5, lane = threadIdx.x & 31;
    int gr = blockIdx.x * 8 + w;           // n*384 + i
    int n = gr / LSEQ, i = gr % LSEQ;
    const float* src = edge + ((size_t)i * LSEQ + n) * DE;   // transpose (0,2,1,3)
    float x[4];
#pragma unroll
    for (int e = 0; e < 4; ++e) x[e] = src[lane*4 + e];
    float s = x[0] + x[1] + x[2] + x[3];
#pragma unroll
    for (int o = 16; o > 0; o >>= 1) s += __shfl_xor(s, o, 32);
    float mu = s * (1.0f / DE);
    float vs = 0.0f;
#pragma unroll
    for (int e = 0; e < 4; ++e) { float d = x[e] - mu; vs += d * d; }
#pragma unroll
    for (int o = 16; o > 0; o >>= 1) vs += __shfl_xor(vs, o, 32);
    float rinv = rsqrtf(vs * (1.0f / DE) + 1e-5f);
    unsigned short* dst = eL + (size_t)gr * DE;
#pragma unroll
    for (int e = 0; e < 4; ++e) {
        int c = lane*4 + e;
        dst[c] = f2b((x[e] - mu) * rinv * g[c] + bta[c]);
    }
}

// ---------------------------------------------------------------------------
// K2: b_h[h][i][j] = LayerNorm(bias[j][i][0:64]) @ Wb.  One wave per (i,j).
// ---------------------------------------------------------------------------
__global__ void k_bias_head(const float* __restrict__ bias, const float* g,
                            const float* bb, const float* Wb, float* bh) {
    int w = threadIdx.x >> 5, lane = threadIdx.x & 31;
    int gr = blockIdx.x * 8 + w;           // i*384 + j
    int i = gr / LSEQ, j = gr % LSEQ;
    const float* src = bias + ((size_t)j * LSEQ + i) * DB;   // bi[i][j] = bias[j][i]
    float x[2];
    x[0] = src[lane*2]; x[1] = src[lane*2 + 1];
    float s = x[0] + x[1];
#pragma unroll
    for (int o = 16; o > 0; o >>= 1) s += __shfl_xor(s, o, 32);
    float mu = s * (1.0f / DB);
    float vs = 0.0f;
#pragma unroll
    for (int e = 0; e < 2; ++e) { float d = x[e] - mu; vs += d * d; }
#pragma unroll
    for (int o = 16; o > 0; o >>= 1) vs += __shfl_xor(vs, o, 32);
    float rinv = rsqrtf(vs * (1.0f / DB) + 1e-5f);
    float ph[NH] = {0.f, 0.f, 0.f, 0.f};
#pragma unroll
    for (int e = 0; e < 2; ++e) {
        int c = lane*2 + e;
        float y = (x[e] - mu) * rinv * g[c] + bb[c];
#pragma unroll
        for (int h = 0; h < NH; ++h) ph[h] += y * Wb[c*NH + h];
    }
#pragma unroll
    for (int h = 0; h < NH; ++h)
#pragma unroll
        for (int o = 16; o > 0; o >>= 1) ph[h] += __shfl_xor(ph[h], o, 32);
    if (lane == 0)
#pragma unroll
        for (int h = 0; h < NH; ++h)
            bh[(size_t)h*NI + (size_t)i*LSEQ + j] = ph[h];
}

// ---------------------------------------------------------------------------
// K3: projection GEMM  [NI,128] x [128,768] via WMMA bf16, scatter epilogue.
// ---------------------------------------------------------------------------
__global__ void __launch_bounds__(256)
k_proj_gemm(const unsigned short* __restrict__ eL,
            const unsigned short* __restrict__ wcatT,
            const float* bg,
            unsigned short* qb, unsigned short* kb,
            unsigned short* vb, unsigned short* gb) {
    int w = threadIdx.x >> 5, lane = threadIdx.x & 31;
    int r0 = blockIdx.x * 16;
    v8f acc[6];
#pragma unroll
    for (int t = 0; t < 6; ++t) acc[t] = zero_v8f();

    for (int k0 = 0; k0 < DE; k0 += 32) {
        Frag a; load_frag(a, eL, DE, r0, k0, lane);
#pragma unroll
        for (int t = 0; t < 6; ++t) {
            int c0 = (w*6 + t) * 16;
            Frag b; load_frag(b, wcatT, DE, c0, k0, lane);   // B^T: [c][k]
            WMMA_BF16(acc[t], a.v, b.v);
        }
    }
#pragma unroll
    for (int t = 0; t < 6; ++t) {
        int c0 = (w*6 + t) * 16;
#pragma unroll
        for (int e = 0; e < 8; ++e) {
            int m  = e + ((lane >= 16) ? 8 : 0);
            int gr = r0 + m, gc = c0 + (lane & 15);
            int n = gr / LSEQ, i = gr % LSEQ;
            int which = gc / DM, hc = gc % DM, h = hc / DH, d = hc % DH;
            float val = acc[t][e];
            if (which == 0)                       // Q: [h][i][n*48+d]
                qb[(size_t)h*HQ + (size_t)i*NK + (size_t)n*DH + d] = f2b(val);
            else if (which == 1)                  // K: [h][j=i][n*48+d]
                kb[(size_t)h*HQ + (size_t)i*NK + (size_t)n*DH + d] = f2b(val);
            else if (which == 2)                  // V^T: [h][n][d][j=i]
                vb[(size_t)h*HV + (size_t)n*DH*LSEQ + (size_t)d*LSEQ + i] = f2b(val);
            else {                                // gate = sigmoid(x + bg)
                float sg = 1.0f / (1.0f + __expf(-(val + bg[hc])));
                gb[((size_t)n*LSEQ + i)*DM + hc] = f2b(sg);
            }
        }
    }
}

// ---------------------------------------------------------------------------
// K4: RoPE (+scale) in-place on Q or K.  One thread per rotation pair.
// ---------------------------------------------------------------------------
__global__ void k_rope_scale(unsigned short* buf, const int* __restrict__ pos,
                             float scale) {
    long t = (long)blockIdx.x * blockDim.x + threadIdx.x;  // < NH*L*L*24
    int d = (int)(t % 24); long r = t / 24;
    int n = (int)(r % LSEQ); r /= LSEQ;
    int row = (int)(r % LSEQ); int h = (int)(r / LSEQ);
    size_t off = (size_t)h*HQ + (size_t)row*NK + (size_t)n*DH + d;
    float x1 = b2f(buf[off]), x2 = b2f(buf[off + 24]);
    // inv_freq = 10000^(-d/24) = exp(-d * ln(10000)/24)
    float ang = (float)pos[row] * __expf(-(float)d * (9.210340371976184f / 24.0f));
    float sn, cs; __sincosf(ang, &sn, &cs);
    buf[off]      = f2b((x1*cs - x2*sn) * scale);
    buf[off + 24] = f2b((x1*sn + x2*cs) * scale);
}

// ---------------------------------------------------------------------------
// K5: per-head scores S = Q K^T (K=18432) + b_h, softmax over j, post-softmax
// mask, store P bf16.  Block = 256 thr: 16 i x 384 j.
// Q strip staged into LDS by the Tensor Data Mover, double-buffered: wave 0
// issues TENSOR_LOAD_TO_LDS for chunk c+1 while all 8 waves compute on chunk
// c; sync = s_wait_tensorcnt + workgroup barrier.  B (K-matrix) streams from
// L2 (113 MB of Q+K resident in the 192 MB L2).
// ---------------------------------------------------------------------------
#define CK 256                       // K-chunk (elements) staged per TDM op
__global__ void __launch_bounds__(256)
k_attn_scores(const unsigned short* __restrict__ qb,
              const unsigned short* __restrict__ kb,
              const float* __restrict__ bh, const float* __restrict__ mask,
              unsigned short* P) {
    __shared__ __align__(16) unsigned short sQ[2][16 * CK];  // 2 x 8 KB
    __shared__ float S[16 * LSEQ];                           // 24 KB
    int w = threadIdx.x >> 5, lane = threadIdx.x & 31;
    int i0 = blockIdx.x * 16, h = blockIdx.y;
    const unsigned short* Q = qb + (size_t)h * HQ;
    const unsigned short* K = kb + (size_t)h * HQ;
    v8f acc[3];
#pragma unroll
    for (int t = 0; t < 3; ++t) acc[t] = zero_v8f();

    const int CH = NK / CK;                                  // 72 chunks
    if (w == 0)                                              // prologue DMA
        tdm_load_2d(lds_addr_of(&sQ[0][0]), Q + (size_t)i0*NK, CK, 16, NK);

    for (int c = 0; c < CH; ++c) {
        if (w == 0) {
            if (c + 1 < CH) {                                // next chunk DMA
                tdm_load_2d(lds_addr_of(&sQ[(c+1) & 1][0]),
                            Q + (size_t)i0*NK + (size_t)(c+1)*CK, CK, 16, NK);
                __builtin_amdgcn_s_wait_tensorcnt(1);        // chunk c landed
            } else {
                __builtin_amdgcn_s_wait_tensorcnt(0);
            }
        }
        __syncthreads();                                     // publish sQ[c&1]
        const unsigned short* sq = &sQ[c & 1][0];
        int kg = c * CK;
        if (kg + 2*CK < NK)
            __builtin_prefetch(K + (size_t)(w*48 + (lane & 15))*NK + kg + 2*CK, 0, 1);
#pragma unroll
        for (int kk = 0; kk < CK; kk += 32) {
            Frag a; load_frag(a, sq, CK, 0, kk, lane);       // ds_load_b128 x2
#pragma unroll
            for (int t = 0; t < 3; ++t) {
                int j0 = (w*3 + t) * 16;
                Frag b; load_frag(b, K, NK, j0, kg + kk, lane);
                WMMA_BF16(acc[t], a.v, b.v);
            }
        }
        __syncthreads();                                     // before overwrite
    }
#pragma unroll
    for (int t = 0; t < 3; ++t) {
        int j0 = (w*3 + t) * 16;
#pragma unroll
        for (int e = 0; e < 8; ++e) {
            int m = e + ((lane >= 16) ? 8 : 0);
            int j = j0 + (lane & 15);
            S[m*LSEQ + j] = acc[t][e] + bh[(size_t)h*NI + (size_t)(i0 + m)*LSEQ + j];
        }
    }
    __syncthreads();
    if (threadIdx.x < 16) {
        int row = threadIdx.x, gi = i0 + row;
        float mx = -3.4e38f;
        for (int j = 0; j < LSEQ; ++j) mx = fmaxf(mx, S[row*LSEQ + j]);
        float sum = 0.0f;
        for (int j = 0; j < LSEQ; ++j) sum += __expf(S[row*LSEQ + j] - mx);
        float inv = 1.0f / sum;
        for (int j = 0; j < LSEQ; ++j) {
            float p = __expf(S[row*LSEQ + j] - mx) * inv
                    + INF_M * (mask[(size_t)gi*LSEQ + j] - 1.0f);
            P[(size_t)h*NI + (size_t)gi*LSEQ + j] = f2b(p);
        }
    }
}

// ---------------------------------------------------------------------------
// K6: out[n,i,h,d] = (sum_j P[h,i,j] * V[h,n,d,j]) * gate.  WMMA, K=384.
// The 48x384 V tile is shared by all waves -> one TDM transfer into LDS,
// then every B fragment is two ds_load_b128.
// ---------------------------------------------------------------------------
__global__ void __launch_bounds__(128)
k_attn_out(const unsigned short* __restrict__ P,
           const unsigned short* __restrict__ vb,
           const unsigned short* __restrict__ gb, unsigned short* ob) {
    __shared__ __align__(16) unsigned short sV[DH * LSEQ];   // 36 KB
    int w = threadIdx.x >> 5, lane = threadIdx.x & 31;
    int itile = blockIdx.x * 4 + w;         // 0..23
    int h = blockIdx.y, n = blockIdx.z;
    int i0 = itile * 16;
    const unsigned short* Ph  = P  + (size_t)h * NI;
    const unsigned short* Vhn = vb + (size_t)h * HV + (size_t)n * DH * LSEQ; // [d][j]

    if (w == 0) {                            // stage V tile once
        tdm_load_2d(lds_addr_of(sV), Vhn, LSEQ, DH, LSEQ);
        __builtin_amdgcn_s_wait_tensorcnt(0);
    }
    __syncthreads();

    v8f acc[3];
#pragma unroll
    for (int t = 0; t < 3; ++t) acc[t] = zero_v8f();

    for (int k0 = 0; k0 < LSEQ; k0 += 32) {
        Frag a; load_frag(a, Ph, LSEQ, i0, k0, lane);
#pragma unroll
        for (int t = 0; t < 3; ++t) {
            Frag b; load_frag(b, sV, LSEQ, t*16, k0, lane);  // rows = d, from LDS
            WMMA_BF16(acc[t], a.v, b.v);
        }
    }
#pragma unroll
    for (int t = 0; t < 3; ++t) {
#pragma unroll
        for (int e = 0; e < 8; ++e) {
            int m = e + ((lane >= 16) ? 8 : 0);
            int i = i0 + m, d = t*16 + (lane & 15);
            size_t idx = ((size_t)n*LSEQ + i)*DM + h*DH + d;
            ob[idx] = f2b(acc[t][e] * b2f(gb[idx]));
        }
    }
}

// ---------------------------------------------------------------------------
// K7: final [NI,192] x Wo[192,128] via WMMA; f32 transposed store to d_out.
// ---------------------------------------------------------------------------
__global__ void __launch_bounds__(256)
k_out_proj(const unsigned short* __restrict__ ob,
           const unsigned short* __restrict__ wobT, float* __restrict__ out) {
    int w = threadIdx.x >> 5, lane = threadIdx.x & 31;
    int r0 = blockIdx.x * 16, c0 = w * 16;
    v8f acc = zero_v8f();
    for (int k0 = 0; k0 < DM; k0 += 32) {
        Frag a; load_frag(a, ob, DM, r0, k0, lane);
        Frag b; load_frag(b, wobT, DM, c0, k0, lane);   // Wo^T: [c][k]
        WMMA_BF16(acc, a.v, b.v);
    }
#pragma unroll
    for (int e = 0; e < 8; ++e) {
        int m  = e + ((lane >= 16) ? 8 : 0);
        int gr = r0 + m;                      // gr = n*384 + i
        int n = gr / LSEQ, i = gr % LSEQ;
        out[((size_t)i*LSEQ + n)*DE + c0 + (lane & 15)] = acc[e];  // un-transpose
    }
}

// ---------------------------------------------------------------------------
extern "C" void kernel_launch(void* const* d_in, const int* in_sizes, int n_in,
                              void* d_out, int out_size, void* d_ws, size_t ws_size,
                              hipStream_t stream) {
    const float* edge     = (const float*)d_in[0];
    const float* bias     = (const float*)d_in[1];
    const int*   edge_pos = (const int*)  d_in[2];
    const float* edge_msk = (const float*)d_in[3];
    const float* ln_e_g   = (const float*)d_in[4];
    const float* ln_e_b   = (const float*)d_in[5];
    const float* ln_b_g   = (const float*)d_in[6];
    const float* ln_b_b   = (const float*)d_in[7];
    const float* Wq       = (const float*)d_in[8];
    const float* Wk       = (const float*)d_in[9];
    const float* Wv       = (const float*)d_in[10];
    const float* Wb       = (const float*)d_in[11];
    const float* Wg       = (const float*)d_in[12];
    const float* bg       = (const float*)d_in[13];
    const float* Wo       = (const float*)d_in[14];
    float* outp           = (float*)d_out;

    char* ws = (char*)d_ws;
    unsigned short* eL    = (unsigned short*)(ws + OFF_EL);
    unsigned short* wcatT = (unsigned short*)(ws + OFF_WCAT);
    unsigned short* wobT  = (unsigned short*)(ws + OFF_WOB);
    unsigned short* qb    = (unsigned short*)(ws + OFF_Q);
    unsigned short* kb    = (unsigned short*)(ws + OFF_K);
    unsigned short* vb    = (unsigned short*)(ws + OFF_V);
    unsigned short* gb    = (unsigned short*)(ws + OFF_G);
    float*          bh    = (float*)         (ws + OFF_BH);
    unsigned short* P     = (unsigned short*)(ws + OFF_P);
    unsigned short* ob    = (unsigned short*)(ws + OFF_O);

    // 0) weights -> bf16, transposed (K contiguous)
    k_prep_weights<<<(768*128 + 128*192 + 255)/256, 256, 0, stream>>>(
        Wq, Wk, Wv, Wg, Wo, wcatT, wobT);
    // 1) LN(edge^T) -> bf16
    k_ln_edge<<<NI/8, 256, 0, stream>>>(edge, ln_e_g, ln_e_b, eL);
    // 2) LN(bias^T) @ Wb -> b_h
    k_bias_head<<<NI/8, 256, 0, stream>>>(bias, ln_b_g, ln_b_b, Wb, bh);
    // 3) QKVG projection (WMMA)
    k_proj_gemm<<<NI/16, 256, 0, stream>>>(eL, wcatT, bg, qb, kb, vb, gb);
    // 4) RoPE + scaling (q * 1/sqrt(dh), k * 1/L)
    {
        long pairs = (long)NH * LSEQ * LSEQ * 24;
        int blocks = (int)((pairs + 255) / 256);
        k_rope_scale<<<blocks, 256, 0, stream>>>(qb, edge_pos, SCALE_Q);
        k_rope_scale<<<blocks, 256, 0, stream>>>(kb, edge_pos, INV_L);
    }
    // 5) scores + softmax + post-softmax mask (WMMA + TDM double buffer)
    k_attn_scores<<<dim3(LSEQ/16, NH), 256, 0, stream>>>(qb, kb, bh, edge_msk, P);
    // 6) attn @ V, gated (WMMA + TDM-staged V tile)
    k_attn_out<<<dim3(6, NH, LSEQ), 128, 0, stream>>>(P, vb, gb, ob);
    // 7) @ Wo + transpose back (WMMA)
    k_out_proj<<<NI/16, 256, 0, stream>>>(ob, wobT, outp);
    (void)in_sizes; (void)n_in; (void)out_size; (void)ws_size;
}
